// GATConv_42915313222057
// MI455X (gfx1250) — compile-verified, matrix operand order checked by reading, not run on hardware
//
#include <hip/hip_runtime.h>
#include <hip/hip_bf16.h>
#include <cstdint>

#define NN 12288
#define CC 64

typedef __attribute__((ext_vector_type(16))) __bf16   v16bf;
typedef __attribute__((ext_vector_type(8)))  float    v8f;
typedef __attribute__((ext_vector_type(8)))  unsigned int v8u;

// Pack two f32 -> one dword of two bf16 (truncation) in ONE v_perm_b32.
// v_perm concat = {S0,S1} with S1 bytes 0-3; sel 2,3 = a[31:16], sel 6,7 = b[31:16].
static __device__ __forceinline__ unsigned int pk_bf16(float a, float b) {
  return __builtin_amdgcn_perm(__float_as_uint(b), __float_as_uint(a), 0x07060302u);
}

// ---------------------------------------------------------------------------
// Kernel 1: wh = x @ W (bf16 WMMA, f32 acc); emit whT (bf16, [64][12288]),
// s[i] = wh[i]·a_src, d[i] = wh[i]·a_dst.  One wave per 16-row block.
// ---------------------------------------------------------------------------
__global__ __launch_bounds__(256) void gat_prep(
    const float* __restrict__ x, const float* __restrict__ W,
    const float* __restrict__ wa, uint16_t* __restrict__ whT,
    float* __restrict__ sv, float* __restrict__ dv)
{
  __shared__ float Wl[CC * CC];
  for (int idx = threadIdx.x * 4; idx < CC * CC; idx += 256 * 4)
    *(float4*)(Wl + idx) = *(const float4*)(W + idx);
  __syncthreads();

  const int lane = threadIdx.x & 31;
  const int wave = threadIdx.x >> 5;
  const int i0   = (blockIdx.x * 8 + wave) * 16;
  const int m    = lane & 15;          // A-row / B-col within tile
  const int koff = (lane >> 4) * 8;    // K sub-group per ISA A/B layout

  v8f acc[4] = {};                      // 4 N-tiles of 16 -> OUT_C = 64
  const float* xr = x + (size_t)(i0 + m) * CC;

  #pragma unroll
  for (int c = 0; c < 2; ++c) {         // K = 64 in two chunks of 32
    const int kb = 32 * c + koff;
    float4 xa0 = *(const float4*)(xr + kb);
    float4 xa1 = *(const float4*)(xr + kb + 4);
    float4 xa2 = *(const float4*)(xr + kb + 16);
    float4 xa3 = *(const float4*)(xr + kb + 20);
    v8u au;
    au[0] = pk_bf16(xa0.x, xa0.y); au[1] = pk_bf16(xa0.z, xa0.w);
    au[2] = pk_bf16(xa1.x, xa1.y); au[3] = pk_bf16(xa1.z, xa1.w);
    au[4] = pk_bf16(xa2.x, xa2.y); au[5] = pk_bf16(xa2.z, xa2.w);
    au[6] = pk_bf16(xa3.x, xa3.y); au[7] = pk_bf16(xa3.z, xa3.w);
    v16bf a = __builtin_bit_cast(v16bf, au);

    #pragma unroll
    for (int t = 0; t < 4; ++t) {
      const int n = 16 * t + m;
      v8u bu;
      #pragma unroll
      for (int h = 0; h < 2; ++h)
        #pragma unroll
        for (int v = 0; v < 4; ++v) {
          const int k = kb + 16 * h + 2 * v;
          bu[4 * h + v] = pk_bf16(Wl[k * CC + n], Wl[(k + 1) * CC + n]);
        }
      v16bf b = __builtin_bit_cast(v16bf, bu);
      acc[t] = __builtin_amdgcn_wmma_f32_16x16x32_bf16(
          false, a, false, b, (short)0, acc[t], false, false);
    }
  }

  // Epilogue: C layout = lanes 0-15 rows 0-7 (VGPR idx), lanes 16-31 rows 8-15.
  const int rb = (lane >> 4) * 8;
  float ps[8] = {}, pd[8] = {};
  #pragma unroll
  for (int t = 0; t < 4; ++t) {
    const int n = 16 * t + m;
    const float as = wa[n];
    const float ad = wa[CC + n];
    #pragma unroll
    for (int r = 0; r < 8; ++r) {
      ps[r] += acc[t][r] * as;
      pd[r] += acc[t][r] * ad;
    }
    // whT[n][i0+rb .. +7] <- acc[t][0..7] : 8 consecutive bf16 = one b128 store
    unsigned int up0 = pk_bf16(acc[t][0], acc[t][1]);
    unsigned int up1 = pk_bf16(acc[t][2], acc[t][3]);
    unsigned int up2 = pk_bf16(acc[t][4], acc[t][5]);
    unsigned int up3 = pk_bf16(acc[t][6], acc[t][7]);
    *(uint4*)(whT + (size_t)n * NN + i0 + rb) = make_uint4(up0, up1, up2, up3);
  }
  // Butterfly reduce s/d over the 16 lanes (columns) of each half-wave.
  #pragma unroll
  for (int ofs = 1; ofs < 16; ofs <<= 1)
    #pragma unroll
    for (int r = 0; r < 8; ++r) {
      ps[r] += __shfl_xor(ps[r], ofs, 16);
      pd[r] += __shfl_xor(pd[r], ofs, 16);
    }
  if (m == 0) {
    #pragma unroll
    for (int r = 0; r < 8; ++r) {
      sv[i0 + rb + r] = ps[r];
      dv[i0 + rb + r] = pd[r];
    }
  }
}

// ---------------------------------------------------------------------------
// Kernel 2: fused masked-softmax attention.  One wave per 16-row block,
// streaming edge_index (604 MB, the HBM bottleneck) in 16x32 tiles.
// acc += exp(leaky(s_i + d_j)) [masked] @ wh  via bf16 WMMA, then divide by
// the row sums and add bias.  No N x N matrix ever touches memory.
// ---------------------------------------------------------------------------
__global__ __launch_bounds__(256) void gat_attn(
    const int* __restrict__ edge, const uint16_t* __restrict__ whT,
    const float* __restrict__ sv, const float* __restrict__ dv,
    const float* __restrict__ bias, float* __restrict__ out)
{
  const int lane = threadIdx.x & 31;
  const int wave = threadIdx.x >> 5;
  const int i0   = (blockIdx.x * 8 + wave) * 16;
  const int m    = lane & 15;
  const int koff = (lane >> 4) * 8;

  const float sm = sv[i0 + m];
  const int* erow = edge + (size_t)(i0 + m) * NN;

  v8f acc[4] = {};
  float lsum = 0.f;

  #pragma unroll 2
  for (int j0 = 0; j0 < NN; j0 += 32) {
    if (j0 + 1024 < NN) __builtin_prefetch(erow + j0 + 1024, 0, 1);
    const int kb = j0 + koff;
    int4   e0 = *(const int4*)(erow + kb);
    int4   e1 = *(const int4*)(erow + kb + 4);
    int4   e2 = *(const int4*)(erow + kb + 16);
    int4   e3 = *(const int4*)(erow + kb + 20);
    float4 d0 = *(const float4*)(dv + kb);
    float4 d1 = *(const float4*)(dv + kb + 4);
    float4 d2 = *(const float4*)(dv + kb + 16);
    float4 d3 = *(const float4*)(dv + kb + 20);

    auto pf = [&](float dval, int ev) -> float {
      float e = sm + dval;
      e = fmaxf(e, 0.01f * e);                 // leaky_relu(0.01): max(e,.01e)
      return (ev > 0) ? __expf(e) : 0.f;       // mask -> exp(-1e9) == 0
    };
    float p[16];
    p[0]  = pf(d0.x, e0.x); p[1]  = pf(d0.y, e0.y);
    p[2]  = pf(d0.z, e0.z); p[3]  = pf(d0.w, e0.w);
    p[4]  = pf(d1.x, e1.x); p[5]  = pf(d1.y, e1.y);
    p[6]  = pf(d1.z, e1.z); p[7]  = pf(d1.w, e1.w);
    p[8]  = pf(d2.x, e2.x); p[9]  = pf(d2.y, e2.y);
    p[10] = pf(d2.z, e2.z); p[11] = pf(d2.w, e2.w);
    p[12] = pf(d3.x, e3.x); p[13] = pf(d3.y, e3.y);
    p[14] = pf(d3.z, e3.z); p[15] = pf(d3.w, e3.w);
    #pragma unroll
    for (int q = 0; q < 16; ++q) lsum += p[q];

    v8u au;
    #pragma unroll
    for (int q = 0; q < 8; ++q) au[q] = pk_bf16(p[2 * q], p[2 * q + 1]);
    v16bf a = __builtin_bit_cast(v16bf, au);

    #pragma unroll
    for (int t = 0; t < 4; ++t) {
      const uint16_t* bp = whT + (size_t)(16 * t + m) * NN + kb;
      uint4 b0 = *(const uint4*)(bp);        // K = kb .. kb+7
      uint4 b1 = *(const uint4*)(bp + 16);   // K = kb+16 .. kb+23
      v8u bu;
      bu[0] = b0.x; bu[1] = b0.y; bu[2] = b0.z; bu[3] = b0.w;
      bu[4] = b1.x; bu[5] = b1.y; bu[6] = b1.z; bu[7] = b1.w;
      v16bf b = __builtin_bit_cast(v16bf, bu);
      acc[t] = __builtin_amdgcn_wmma_f32_16x16x32_bf16(
          false, a, false, b, (short)0, acc[t], false, false);
    }
  }

  // Row sums: lane L and L+16 each hold half of row (L&15).
  const float lfull = lsum + __shfl_xor(lsum, 16, 32);
  const int rb = (lane >> 4) * 8;
  float linv[8];
  #pragma unroll
  for (int r = 0; r < 8; ++r) {
    float lv = __shfl(lfull, rb + r, 32);    // lane (rb+r) holds row rb+r
    linv[r] = 1.0f / fmaxf(lv, 1e-30f);
  }
  #pragma unroll
  for (int t = 0; t < 4; ++t) {
    const int n = 16 * t + m;
    const float bv = bias[n];
    #pragma unroll
    for (int r = 0; r < 8; ++r)
      out[(size_t)(i0 + rb + r) * CC + n] = acc[t][r] * linv[r] + bv;
  }
}

// ---------------------------------------------------------------------------
extern "C" void kernel_launch(void* const* d_in, const int* in_sizes, int n_in,
                              void* d_out, int out_size, void* d_ws, size_t ws_size,
                              hipStream_t stream) {
  const float* x    = (const float*)d_in[0];
  const int*   edge = (const int*)  d_in[1];
  const float* W    = (const float*)d_in[2];
  const float* wa   = (const float*)d_in[3];
  const float* bias = (const float*)d_in[4];
  float* out = (float*)d_out;

  char* ws = (char*)d_ws;
  uint16_t* whT = (uint16_t*)ws;                       // 64*12288 bf16 = 1.5 MB
  float* sv = (float*)(ws + (size_t)CC * NN * 2);      // 12288 f32
  float* dv = sv + NN;                                 // 12288 f32

  gat_prep<<<dim3(96), dim3(256), 0, stream>>>(x, W, wa, whT, sv, dv);
  gat_attn<<<dim3(96), dim3(256), 0, stream>>>(edge, whT, sv, dv, bias, out);
}
